// MambaBody_83932250898892
// MI455X (gfx1250) — matrix operand order repaired and checked by reading, not compiled, not run on hardware
//
#include <hip/hip_runtime.h>
#include <stdint.h>

// ---- problem constants (match reference) ----
#define D_MODEL 1024
#define D_STATE 16
#define N_LAYERS 2
#define D_INNER 2048
#define DT_RANK 64
#define D_CONV 4
#define BB 2
#define LL 1024
#define NTOK (BB * LL)                   // 2048 tokens
#define XPROJ_N (DT_RANK + 2 * D_STATE)  // 96

typedef unsigned short u16;
typedef __attribute__((ext_vector_type(16))) __bf16 v16bf;
typedef __attribute__((ext_vector_type(8)))  float  v8f;
typedef __attribute__((ext_vector_type(4)))  unsigned int v4u;

// CDNA5 async global->LDS path (ASYNCcnt-tracked), with safe fallback.
#if __has_builtin(__builtin_amdgcn_global_load_async_to_lds_b32) && \
    __has_builtin(__builtin_amdgcn_s_wait_asynccnt)
#define HAVE_ASYNC_LDS 1
#endif

__device__ __forceinline__ u16 f2bf(float f) {
  union { float f; uint32_t u; } c; c.f = f;
  uint32_t u = c.u;
  uint32_t r = u + 0x7FFFu + ((u >> 16) & 1u);   // round-to-nearest-even
  return (u16)(r >> 16);
}

__device__ __forceinline__ float silu_f(float x) {
  return x / (1.0f + __expf(-x));
}

__device__ __forceinline__ float softplus_f(float x) {
  return (x > 20.0f) ? x : log1pf(__expf(x));
}

// A fragment (16x32 bf16): halfword i -> K = 16*hi + i (one contiguous 32B load)
__device__ __forceinline__ v16bf load_a_frag(const u16* row, int kb) {
  return *(const v16bf*)(row + kb);
}
// B fragment (32x16 bf16): halfword i -> K = 16*hi + i (one contiguous 32B load)
__device__ __forceinline__ v16bf load_b_frag(const u16* row, int k) {
  return *(const v16bf*)(row + k);
}

// ---------------- fp32 -> bf16 cast ----------------
__global__ void cast_bf16_kernel(u16* __restrict__ dst, const float* __restrict__ src, int n) {
  int i = blockIdx.x * blockDim.x + threadIdx.x;
  if (i < n) dst[i] = f2bf(src[i]);
}

// ---------------- LayerNorm over D_MODEL, one block (256 thr) per token ----------------
__global__ void layernorm_kernel(const float* __restrict__ x,
                                 const float* __restrict__ g,
                                 const float* __restrict__ b,
                                 u16* __restrict__ outbf,
                                 float* __restrict__ outf,
                                 int write_bf) {
  __shared__ float s1[256], s2[256];
  int row = blockIdx.x, t = threadIdx.x;
  const float4* xr = (const float4*)(x + (size_t)row * D_MODEL);
  float4 v = xr[t];
  s1[t] = v.x + v.y + v.z + v.w;
  s2[t] = v.x * v.x + v.y * v.y + v.z * v.z + v.w * v.w;
  __syncthreads();
  for (int o = 128; o > 0; o >>= 1) {
    if (t < o) { s1[t] += s1[t + o]; s2[t] += s2[t + o]; }
    __syncthreads();
  }
  float mean = s1[0] * (1.0f / D_MODEL);
  float var  = s2[0] * (1.0f / D_MODEL) - mean * mean;
  float rstd = rsqrtf(var + 1e-5f);
  float4 gg = ((const float4*)g)[t];
  float4 bv = ((const float4*)b)[t];
  float o0 = (v.x - mean) * rstd * gg.x + bv.x;
  float o1 = (v.y - mean) * rstd * gg.y + bv.y;
  float o2 = (v.z - mean) * rstd * gg.z + bv.z;
  float o3 = (v.w - mean) * rstd * gg.w + bv.w;
  size_t base = (size_t)row * D_MODEL + t * 4;
  if (write_bf) {
    outbf[base + 0] = f2bf(o0); outbf[base + 1] = f2bf(o1);
    outbf[base + 2] = f2bf(o2); outbf[base + 3] = f2bf(o3);
  } else {
    outf[base + 0] = o0; outf[base + 1] = o1;
    outf[base + 2] = o2; outf[base + 3] = o3;
  }
}

// ---------------- epilogue helper ----------------
// mode 0: C = acc ; mode 1: C += acc ; mode 2: C = softplus(acc + bias[n])
__device__ __forceinline__ void store_tile(float* __restrict__ C, int N,
                                           int m0, int n0, int lo, int hi,
                                           v8f acc, const float* __restrict__ bias,
                                           int mode) {
  float* Cc = C + (size_t)(m0 + (hi << 3)) * N + n0 + lo;
  if (mode == 1) {
#pragma unroll
    for (int r = 0; r < 8; ++r) Cc[(size_t)r * N] += acc[r];
  } else if (mode == 2) {
    float bb = bias[n0 + lo];
#pragma unroll
    for (int r = 0; r < 8; ++r) Cc[(size_t)r * N] = softplus_f(acc[r] + bb);
  } else {
#pragma unroll
    for (int r = 0; r < 8; ++r) Cc[(size_t)r * N] = acc[r];
  }
}

// ---------------- blocked BF16 WMMA GEMM: C[M,N] = A[M,K] * W[N,K]^T ----------------
// Each wave: 2x2 WMMA tiles (32M x 32N) -> 4 WMMAs per 8 b128 loads.
// Block = 8 waves as 2(M) x 4(N) -> 64M x 128N. M % 64 == 0, N % 128 == 0.
__global__ void gemm_wmma_bf16_blocked(const u16* __restrict__ A,
                                       const u16* __restrict__ W,
                                       float* __restrict__ C,
                                       const float* __restrict__ bias,
                                       int M, int N, int K, int mode) {
  int lane = threadIdx.x & 31;
  int wave = threadIdx.x >> 5;
  int wm = wave >> 2;                         // 0..1
  int wn = wave & 3;                          // 0..3
  int m0 = (blockIdx.x << 6) + (wm << 5);     // wave covers rows m0..m0+31
  int n0 = (blockIdx.y << 7) + (wn << 5);     // wave covers cols n0..n0+31
  int lo = lane & 15, hi = lane >> 4;

  const u16* A0 = A + (size_t)(m0 + lo) * K;
  const u16* A1 = A0 + (size_t)16 * K;
  const u16* W0 = W + (size_t)(n0 + lo) * K;
  const u16* W1 = W0 + (size_t)16 * K;

  v8f z = {0.f, 0.f, 0.f, 0.f, 0.f, 0.f, 0.f, 0.f};
  v8f acc00 = z, acc01 = z, acc10 = z, acc11 = z;

  for (int k0 = 0; k0 < K; k0 += 32) {
    int kb = k0 + (hi << 4);
    int kbb = k0 + (hi << 4);
    v16bf a0 = load_a_frag(A0, kb);
    v16bf a1 = load_a_frag(A1, kb);
    v16bf b0 = load_b_frag(W0, kbb);
    v16bf b1 = load_b_frag(W1, kbb);
    acc00 = __builtin_amdgcn_wmma_f32_16x16x32_bf16(false, a0, false, b0, (short)0, acc00, false, false);
    acc01 = __builtin_amdgcn_wmma_f32_16x16x32_bf16(false, a0, false, b1, (short)0, acc01, false, false);
    acc10 = __builtin_amdgcn_wmma_f32_16x16x32_bf16(false, a1, false, b0, (short)0, acc10, false, false);
    acc11 = __builtin_amdgcn_wmma_f32_16x16x32_bf16(false, a1, false, b1, (short)0, acc11, false, false);
  }

  store_tile(C, N, m0,      n0,      lo, hi, acc00, bias, mode);
  store_tile(C, N, m0,      n0 + 16, lo, hi, acc01, bias, mode);
  store_tile(C, N, m0 + 16, n0,      lo, hi, acc10, bias, mode);
  store_tile(C, N, m0 + 16, n0 + 16, lo, hi, acc11, bias, mode);
}

// ---------------- simple 1-tile WMMA GEMM for x_proj (N=96) ----------------
// Also emits the dt slice (cols < DT_RANK) as packed bf16 for the dt GEMM.
__global__ void gemm_wmma_bf16_simple(const u16* __restrict__ A,
                                      const u16* __restrict__ W,
                                      float* __restrict__ C,
                                      u16* __restrict__ dtbf,
                                      int M, int N, int K) {
  int lane  = threadIdx.x & 31;
  int wave  = threadIdx.x >> 5;
  int mtile = blockIdx.x << 4;
  int ntile = (blockIdx.y << 7) + (wave << 4);
  if (ntile >= N) return;                     // wave-uniform, EXEC stays all-1s
  int lo = lane & 15, hi = lane >> 4;

  const u16* Arow = A + (size_t)(mtile + lo) * K;
  const u16* Wrow = W + (size_t)(ntile + lo) * K;

  v8f acc = {0.f, 0.f, 0.f, 0.f, 0.f, 0.f, 0.f, 0.f};
  for (int k0 = 0; k0 < K; k0 += 32) {
    v16bf a = load_a_frag(Arow, k0 + (hi << 4));
    v16bf b = load_b_frag(Wrow, k0 + (hi << 4));
    acc = __builtin_amdgcn_wmma_f32_16x16x32_bf16(false, a, false, b, (short)0, acc, false, false);
  }
  store_tile(C, N, mtile, ntile, lo, hi, acc, nullptr, 0);
  if (dtbf != nullptr && ntile < DT_RANK) {   // fused dt extraction (bf16)
    u16* Dd = dtbf + (size_t)(mtile + (hi << 3)) * DT_RANK + ntile + lo;
#pragma unroll
    for (int r = 0; r < 8; ++r) Dd[(size_t)r * DT_RANK] = f2bf(acc[r]);
  }
}

// ---------------- causal depthwise conv(4) + bias + SiLU ----------------
__global__ void conv_silu_kernel(const float* __restrict__ xz,
                                 const float* __restrict__ cw,
                                 const float* __restrict__ cb,
                                 float* __restrict__ u,
                                 u16* __restrict__ ubf) {
  int i = blockIdx.x * blockDim.x + threadIdx.x;
  if (i >= NTOK * D_INNER) return;
  int r = i >> 11;            // token index (b*L + l)
  int d = i & (D_INNER - 1);
  int l = r & (LL - 1);
  const float* w4 = cw + (size_t)d * D_CONV;
  float s = cb[d];
#pragma unroll
  for (int k = 0; k < D_CONV; ++k) {
    int lk = l + k - (D_CONV - 1);
    if (lk >= 0)
      s += w4[k] * xz[(size_t)(r + k - (D_CONV - 1)) * (2 * D_INNER) + d];
  }
  float v = silu_f(s);
  u[i] = v;
  ubf[i] = f2bf(v);
}

// ---------------- selective scan, fused with output gate ----------------
// grid = B * (D_INNER/256) = 16 blocks, 256 threads; one thread per (b,d).
// 16-wide state in registers; B_t/C_t staged via LDS in 8-step chunks using
// CDNA5 async global->LDS copies when available (ASYNCcnt-tracked).
__global__ void scan_gate_kernel(const float* __restrict__ delta,
                                 const float* __restrict__ u,
                                 const float* __restrict__ xdbl,
                                 const float* __restrict__ A_log,
                                 const float* __restrict__ xz,
                                 const float* __restrict__ Dp,
                                 u16* __restrict__ ybf) {
  __shared__ float sBC[8][32];   // [step][0..15]=B_t, [16..31]=C_t
  int b = blockIdx.x >> 3;
  int d = ((blockIdx.x & 7) << 8) + threadIdx.x;
  float Av[D_STATE], h[D_STATE];
#pragma unroll
  for (int s = 0; s < D_STATE; ++s) {
    Av[s] = -__expf(A_log[(size_t)d * D_STATE + s]);
    h[s] = 0.0f;
  }
  float Dpd = Dp[d];
  int t = threadIdx.x;
  int ls = t >> 5, lj = t & 31;   // 8 steps x 32 values per chunk
  for (int l0 = 0; l0 < LL; l0 += 8) {
    size_t gidx = (size_t)(b * LL + l0 + ls) * XPROJ_N + DT_RANK + lj;
#ifdef HAVE_ASYNC_LDS
    __syncthreads();              // previous chunk fully consumed
    __builtin_amdgcn_global_load_async_to_lds_b32(
        (__attribute__((address_space(1))) int*)(xdbl + gidx),
        (__attribute__((address_space(3))) int*)(&sBC[ls][lj]),
        0, 0);
    __builtin_amdgcn_s_wait_asynccnt(0);
    __syncthreads();              // all waves' async copies visible
#else
    float v = xdbl[gidx];
    __syncthreads();              // previous chunk fully consumed
    sBC[ls][lj] = v;
    __syncthreads();
#endif
#pragma unroll
    for (int st = 0; st < 8; ++st) {
      int r = b * LL + l0 + st;
      float dt = delta[(size_t)r * D_INNER + d];
      float ut = u[(size_t)r * D_INNER + d];
      float du = dt * ut;
      float y = 0.0f;
#pragma unroll
      for (int s = 0; s < D_STATE; ++s) {
        float hs = __expf(dt * Av[s]) * h[s] + du * sBC[st][s];
        h[s] = hs;
        y += hs * sBC[st][16 + s];
      }
      float res = xz[(size_t)r * (2 * D_INNER) + D_INNER + d];
      ybf[(size_t)r * D_INNER + d] = f2bf((y + ut * Dpd) * silu_f(res));
    }
  }
}

// ---------------- host-side orchestration ----------------
extern "C" void kernel_launch(void* const* d_in, const int* in_sizes, int n_in,
                              void* d_out, int out_size, void* d_ws, size_t ws_size,
                              hipStream_t stream) {
  const float* x      = (const float*)d_in[0];
  const float* ln_g   = (const float*)d_in[1];
  const float* ln_b   = (const float*)d_in[2];
  const float* in_w   = (const float*)d_in[3];
  const float* conv_w = (const float*)d_in[4];
  const float* conv_b = (const float*)d_in[5];
  const float* xp_w   = (const float*)d_in[6];
  const float* dt_w   = (const float*)d_in[7];
  const float* dt_b   = (const float*)d_in[8];
  const float* A_log  = (const float*)d_in[9];
  const float* Dp     = (const float*)d_in[10];
  const float* out_w  = (const float*)d_in[11];
  const float* fn_g   = (const float*)d_in[12];
  const float* fn_b   = (const float*)d_in[13];

  char* ws = (char*)d_ws;
  size_t off = 0;
  auto alloc = [&](size_t bytes) {
    void* p = ws + off;
    off = (off + bytes + 255) & ~(size_t)255;
    return p;
  };

  u16*   wbf_in  = (u16*)alloc((size_t)N_LAYERS * 2 * D_INNER * D_MODEL * 2);
  u16*   wbf_xp  = (u16*)alloc((size_t)N_LAYERS * XPROJ_N * D_INNER * 2);
  u16*   wbf_dt  = (u16*)alloc((size_t)N_LAYERS * D_INNER * DT_RANK * 2);
  u16*   wbf_out = (u16*)alloc((size_t)N_LAYERS * D_MODEL * D_INNER * 2);
  float* xcur    = (float*)alloc((size_t)NTOK * D_MODEL * 4);
  u16*   xnbf    = (u16*)alloc((size_t)NTOK * D_MODEL * 2);
  float* xz      = (float*)alloc((size_t)NTOK * 2 * D_INNER * 4);
  float* ubuf    = (float*)alloc((size_t)NTOK * D_INNER * 4);
  u16*   ubf     = (u16*)alloc((size_t)NTOK * D_INNER * 2);
  float* xdbl    = (float*)alloc((size_t)NTOK * XPROJ_N * 4);
  u16*   dtbf    = (u16*)alloc((size_t)NTOK * DT_RANK * 2);
  float* delta   = (float*)alloc((size_t)NTOK * D_INNER * 4);
  u16*   ybf     = (u16*)alloc((size_t)NTOK * D_INNER * 2);

  auto cast = [&](u16* dst, const float* src, size_t n) {
    cast_bf16_kernel<<<(unsigned)((n + 255) / 256), 256, 0, stream>>>(dst, src, (int)n);
  };
  // blocked GEMM: M%64==0, N%128==0
  auto gemm = [&](const u16* A, const u16* W, float* C, int M, int N, int K,
                  const float* bias, int mode) {
    dim3 g((unsigned)(M / 64), (unsigned)(N / 128));
    gemm_wmma_bf16_blocked<<<g, 256, 0, stream>>>(A, W, C, bias, M, N, K, mode);
  };

  // per-call weight casts (deterministic; graph-capture safe)
  cast(wbf_in,  in_w,  (size_t)N_LAYERS * 2 * D_INNER * D_MODEL);
  cast(wbf_xp,  xp_w,  (size_t)N_LAYERS * XPROJ_N * D_INNER);
  cast(wbf_dt,  dt_w,  (size_t)N_LAYERS * D_INNER * DT_RANK);
  cast(wbf_out, out_w, (size_t)N_LAYERS * D_MODEL * D_INNER);

  // residual stream lives in workspace (never mutate d_in)
  (void)hipMemcpyAsync(xcur, x, (size_t)NTOK * D_MODEL * sizeof(float),
                       hipMemcpyDeviceToDevice, stream);

  const int nInner = NTOK * D_INNER;
  const unsigned gInner = (unsigned)((nInner + 255) / 256);

  for (int i = 0; i < N_LAYERS; ++i) {
    // 1. LayerNorm -> bf16 activations
    layernorm_kernel<<<NTOK, 256, 0, stream>>>(
        xcur, ln_g + (size_t)i * D_MODEL, ln_b + (size_t)i * D_MODEL,
        xnbf, nullptr, 1);
    // 2. in_proj: xz = xn @ in_w^T   [2048 x 4096]
    gemm(xnbf, wbf_in + (size_t)i * 2 * D_INNER * D_MODEL, xz,
         NTOK, 2 * D_INNER, D_MODEL, nullptr, 0);
    // 3. causal depthwise conv + SiLU -> u (fp32 + bf16)
    conv_silu_kernel<<<gInner, 256, 0, stream>>>(
        xz, conv_w + (size_t)i * D_INNER * D_CONV, conv_b + (size_t)i * D_INNER,
        ubuf, ubf);
    // 4. x_proj: xdbl = u @ xpw^T  [2048 x 96], dt slice fused out as bf16
    {
      dim3 g((unsigned)(NTOK / 16), (unsigned)((XPROJ_N + 127) / 128));
      gemm_wmma_bf16_simple<<<g, 256, 0, stream>>>(
          ubf, wbf_xp + (size_t)i * XPROJ_N * D_INNER, xdbl, dtbf,
          NTOK, XPROJ_N, D_INNER);
    }
    // 5. dt_proj with fused bias+softplus epilogue: delta = softplus(dt@dtw^T + dtb)
    gemm(dtbf, wbf_dt + (size_t)i * D_INNER * DT_RANK, delta,
         NTOK, D_INNER, DT_RANK, dt_b + (size_t)i * D_INNER, 2);
    // 6. selective scan fused with gate -> ybf (bf16)
    scan_gate_kernel<<<BB * (D_INNER / 256), 256, 0, stream>>>(
        delta, ubuf, xdbl, A_log + (size_t)i * D_INNER * D_STATE,
        xz, Dp + (size_t)i * D_INNER, ybf);
    // 7. out_proj with residual accumulate: xcur += y @ ow^T
    gemm(ybf, wbf_out + (size_t)i * D_MODEL * D_INNER, xcur,
         NTOK, D_MODEL, D_INNER, nullptr, 1);
  }

  // final LayerNorm -> fp32 output
  layernorm_kernel<<<NTOK, 256, 0, stream>>>(xcur, fn_g, fn_b,
                                             nullptr, (float*)d_out, 0);
}
